// GraphSAGE_75479755259981
// MI455X (gfx1250) — matrix operand order repaired and checked by reading, not compiled, not run on hardware
//
#include <hip/hip_runtime.h>
#include <hip/hip_bf16.h>

// ---------------------------------------------------------------------------
// GraphSAGE 2-layer forward for MI455X (gfx1250, wave32).
//   layer: agg = segment_mean(x[src] -> dst); out = agg@Wl^T + b + x@Wr^T;
//          out = l2norm(out); (relu for layer 1); final: l2norm + log_softmax
// GEMMs run on v_wmma_f32_16x16x32_bf16 (fp32 accumulate); aggregation uses
// hardware fp32 global atomics (L2-resident working set, 192MB L2).
// ---------------------------------------------------------------------------

typedef __attribute__((ext_vector_type(16))) __bf16 v16bf;
typedef __attribute__((ext_vector_type(8)))  __bf16 bf16x8;
typedef __attribute__((ext_vector_type(8)))  float  v8f;

union Frag16 { v16bf v; bf16x8 h[2]; };

__device__ __forceinline__ v8f wmma_bf16(v16bf a, v16bf b, v8f c) {
    // 8 args: (neg_a, A, neg_b, B, c_mod, C, reuse_a, reuse_b)
    return __builtin_amdgcn_wmma_f32_16x16x32_bf16(false, a, false, b,
                                                   (short)0, c, false, false);
}

#define LDA 136  // LDS row stride in bf16: 272B -> conflict-free ds_load_b128

// ---------------------------------------------------------------------------
// zero fill
__global__ void zero_kernel(float* __restrict__ p, long long n) {
    long long i = ((long long)blockIdx.x * blockDim.x + threadIdx.x) * 4;
    if (i + 3 < n) {
        *(float4*)(p + i) = make_float4(0.f, 0.f, 0.f, 0.f);
    } else {
        for (long long j = i; j < n; ++j) p[j] = 0.f;
    }
}

// ---------------------------------------------------------------------------
// convert weights to bf16 (W2 zero-padded from 40 to 48 output rows)
__global__ void prep_weights(const float* __restrict__ W1l, const float* __restrict__ W1r,
                             const float* __restrict__ W2l, const float* __restrict__ W2r,
                             __bf16* __restrict__ W1l_bf, __bf16* __restrict__ W1r_bf,
                             __bf16* __restrict__ W2l_bf, __bf16* __restrict__ W2r_bf) {
    int i = blockIdx.x * blockDim.x + threadIdx.x;
    if (i < 128 * 128) {
        W1l_bf[i] = (__bf16)W1l[i];
        W1r_bf[i] = (__bf16)W1r[i];
    }
    if (i < 48 * 128) {
        int r = i >> 7;
        W2l_bf[i] = (__bf16)((r < 40) ? W2l[i] : 0.f);
        W2r_bf[i] = (__bf16)((r < 40) ? W2r[i] : 0.f);
    }
}

// ---------------------------------------------------------------------------
// mean-aggregation scatter: one wave per edge; 512B coalesced gather,
// 128 lane-parallel fp32 global atomics into agg[dst].
__global__ __launch_bounds__(256)
void scatter_kernel(const float* __restrict__ feat, const long long* __restrict__ ei,
                    long long E, float* __restrict__ agg, float* __restrict__ cnt,
                    int addCnt) {
    long long wid = (long long)blockIdx.x * 8 + (threadIdx.x >> 5);
    int lane = threadIdx.x & 31;
    if (wid >= E) return;
    long long s = ei[wid];
    long long d = ei[E + wid];
    float4 v = *(const float4*)(feat + s * 128 + lane * 4);
    float* base = agg + d * 128 + lane * 4;
    unsafeAtomicAdd(base + 0, v.x);
    unsafeAtomicAdd(base + 1, v.y);
    unsafeAtomicAdd(base + 2, v.z);
    unsafeAtomicAdd(base + 3, v.w);
    if (addCnt && lane == 0) unsafeAtomicAdd(cnt + d, 1.0f);
}

// ---------------------------------------------------------------------------
// layer 1: h = relu(l2norm((agg/cnt)@Wl^T + b + x@Wr^T))
// 256 threads = 8 waves; block tile = 128 rows x 128 cols; wave = one 16-col
// tile over 8 row tiles; K=128 in 4 steps of 32 (bf16 WMMA).
__global__ __launch_bounds__(256)
void gemm1_kernel(const float* __restrict__ agg, const float* __restrict__ cnt,
                  const float* __restrict__ x,
                  const __bf16* __restrict__ Wl, const __bf16* __restrict__ Wr,
                  const float* __restrict__ b1, float* __restrict__ h, int N) {
    __shared__ __bf16 As[128][LDA];   // normalized aggregation (bf16)
    __shared__ __bf16 Xs[128][LDA];   // root features (bf16)
    __shared__ float  sumsq[128];

    const int tid = threadIdx.x;
    const int rowbase = blockIdx.x * 128;
    if (tid < 128) sumsq[tid] = 0.0f;

    // cooperative staging, float4 granularity, with mean normalization
    for (int i = tid; i < 128 * 32; i += 256) {
        int r = i >> 5;
        int c = (i & 31) << 2;
        int gr = rowbase + r;
        float4 va = make_float4(0.f, 0.f, 0.f, 0.f), vx = va;
        float scale = 0.f;
        if (gr < N) {
            scale = 1.0f / fmaxf(cnt[gr], 1.0f);
            va = *(const float4*)(agg + (size_t)gr * 128 + c);
            vx = *(const float4*)(x   + (size_t)gr * 128 + c);
        }
        As[r][c + 0] = (__bf16)(va.x * scale);
        As[r][c + 1] = (__bf16)(va.y * scale);
        As[r][c + 2] = (__bf16)(va.z * scale);
        As[r][c + 3] = (__bf16)(va.w * scale);
        Xs[r][c + 0] = (__bf16)vx.x;
        Xs[r][c + 1] = (__bf16)vx.y;
        Xs[r][c + 2] = (__bf16)vx.z;
        Xs[r][c + 3] = (__bf16)vx.w;
    }
    __syncthreads();

    const int wave = tid >> 5;
    const int lane = tid & 31;
    const int jl   = lane & 15;        // column within tile (B/C layout)
    const int hi   = lane >> 4;        // half-wave select
    const int col  = wave * 16 + jl;

    v8f acc[8] = {};
    #pragma unroll
    for (int kk = 0; kk < 4; ++kk) {
        const int k0 = kk * 32;
        // B fragments: 32x16 bf16; lane j holds col j, K 0-7,16-23 (lo half) / 8-15,24-31 (hi)
        Frag16 bl, br;
        const __bf16* wlp = Wl + (size_t)col * 128 + k0 + hi * 8;
        const __bf16* wrp = Wr + (size_t)col * 128 + k0 + hi * 8;
        bl.h[0] = *(const bf16x8*)(wlp);  bl.h[1] = *(const bf16x8*)(wlp + 16);
        br.h[0] = *(const bf16x8*)(wrp);  br.h[1] = *(const bf16x8*)(wrp + 16);
        #pragma unroll
        for (int rt = 0; rt < 8; ++rt) {
            // A fragments: 16x32 bf16; row = lane&15; K 0-7,16-23 (lo) / 8-15,24-31 (hi)
            Frag16 aa, ax;
            const __bf16* ap = &As[rt * 16 + jl][k0 + hi * 8];
            const __bf16* xp = &Xs[rt * 16 + jl][k0 + hi * 8];
            aa.h[0] = *(const bf16x8*)(ap);  aa.h[1] = *(const bf16x8*)(ap + 16);
            ax.h[0] = *(const bf16x8*)(xp);  ax.h[1] = *(const bf16x8*)(xp + 16);
            acc[rt] = wmma_bf16(aa.v, bl.v, acc[rt]);
            acc[rt] = wmma_bf16(ax.v, br.v, acc[rt]);
        }
    }

    // bias, then per-row sum of squares across all 8 waves via LDS atomics
    const float bias = b1[col];
    #pragma unroll
    for (int rt = 0; rt < 8; ++rt) {
        #pragma unroll
        for (int i = 0; i < 8; ++i) {
            float v = acc[rt][i] + bias;   // C/D: vgpr i -> row i (+8 for hi half)
            acc[rt][i] = v;
            atomicAdd(&sumsq[rt * 16 + i + hi * 8], v * v);
        }
    }
    __syncthreads();

    #pragma unroll
    for (int rt = 0; rt < 8; ++rt) {
        #pragma unroll
        for (int i = 0; i < 8; ++i) {
            int r  = rt * 16 + i + hi * 8;
            int gr = rowbase + r;
            if (gr < N) {
                float inv = 1.0f / fmaxf(sqrtf(sumsq[r]), 1e-12f);
                h[(size_t)gr * 128 + col] = fmaxf(acc[rt][i] * inv, 0.0f);
            }
        }
    }
}

// ---------------------------------------------------------------------------
// layer 2: raw logits out[n,c] = (agg/cnt)@W2l^T + b2 + h@W2r^T  (c < 40)
// 96 threads = 3 waves (48 padded cols); norm + log_softmax deferred.
__global__ __launch_bounds__(96)
void gemm2_kernel(const float* __restrict__ agg, const float* __restrict__ cnt,
                  const float* __restrict__ hfeat,
                  const __bf16* __restrict__ Wl, const __bf16* __restrict__ Wr,
                  const float* __restrict__ b2, float* __restrict__ out, int N) {
    __shared__ __bf16 As[128][LDA];
    __shared__ __bf16 Hs[128][LDA];

    const int tid = threadIdx.x;
    const int rowbase = blockIdx.x * 128;

    for (int i = tid; i < 128 * 32; i += 96) {
        int r = i >> 5;
        int c = (i & 31) << 2;
        int gr = rowbase + r;
        float4 va = make_float4(0.f, 0.f, 0.f, 0.f), vh = va;
        float scale = 0.f;
        if (gr < N) {
            scale = 1.0f / fmaxf(cnt[gr], 1.0f);
            va = *(const float4*)(agg   + (size_t)gr * 128 + c);
            vh = *(const float4*)(hfeat + (size_t)gr * 128 + c);
        }
        As[r][c + 0] = (__bf16)(va.x * scale);
        As[r][c + 1] = (__bf16)(va.y * scale);
        As[r][c + 2] = (__bf16)(va.z * scale);
        As[r][c + 3] = (__bf16)(va.w * scale);
        Hs[r][c + 0] = (__bf16)vh.x;
        Hs[r][c + 1] = (__bf16)vh.y;
        Hs[r][c + 2] = (__bf16)vh.z;
        Hs[r][c + 3] = (__bf16)vh.w;
    }
    __syncthreads();

    const int wave = tid >> 5;
    const int lane = tid & 31;
    const int jl   = lane & 15;
    const int hi   = lane >> 4;
    const int col  = wave * 16 + jl;      // 0..47 (>=40 are zero-padded)

    v8f acc[8] = {};
    #pragma unroll
    for (int kk = 0; kk < 4; ++kk) {
        const int k0 = kk * 32;
        Frag16 bl, br;
        const __bf16* wlp = Wl + (size_t)col * 128 + k0 + hi * 8;
        const __bf16* wrp = Wr + (size_t)col * 128 + k0 + hi * 8;
        bl.h[0] = *(const bf16x8*)(wlp);  bl.h[1] = *(const bf16x8*)(wlp + 16);
        br.h[0] = *(const bf16x8*)(wrp);  br.h[1] = *(const bf16x8*)(wrp + 16);
        #pragma unroll
        for (int rt = 0; rt < 8; ++rt) {
            Frag16 aa, ah;
            const __bf16* ap = &As[rt * 16 + jl][k0 + hi * 8];
            const __bf16* hp = &Hs[rt * 16 + jl][k0 + hi * 8];
            aa.h[0] = *(const bf16x8*)(ap);  aa.h[1] = *(const bf16x8*)(ap + 16);
            ah.h[0] = *(const bf16x8*)(hp);  ah.h[1] = *(const bf16x8*)(hp + 16);
            acc[rt] = wmma_bf16(aa.v, bl.v, acc[rt]);
            acc[rt] = wmma_bf16(ah.v, br.v, acc[rt]);
        }
    }

    if (col < 40) {
        const float bias = b2[col];
        #pragma unroll
        for (int rt = 0; rt < 8; ++rt) {
            #pragma unroll
            for (int i = 0; i < 8; ++i) {
                int gr = rowbase + rt * 16 + i + hi * 8;
                if (gr < N) out[(size_t)gr * 40 + col] = acc[rt][i] + bias;
            }
        }
    }
}

// ---------------------------------------------------------------------------
// per-row: L2 normalize 40 logits, then log_softmax; in-place on d_out.
__global__ __launch_bounds__(256)
void finalize_kernel(float* __restrict__ out, int N) {
    int n = blockIdx.x * blockDim.x + threadIdx.x;
    if (n >= N) return;
    float v[40];
    float ss = 0.f;
    #pragma unroll
    for (int c = 0; c < 40; ++c) { v[c] = out[(size_t)n * 40 + c]; ss += v[c] * v[c]; }
    float inv = 1.0f / fmaxf(sqrtf(ss), 1e-12f);
    float m = -3.402823e38f;
    #pragma unroll
    for (int c = 0; c < 40; ++c) { v[c] *= inv; m = fmaxf(m, v[c]); }
    float s = 0.f;
    #pragma unroll
    for (int c = 0; c < 40; ++c) s += expf(v[c] - m);
    float lse = m + logf(s);
    #pragma unroll
    for (int c = 0; c < 40; ++c) out[(size_t)n * 40 + c] = v[c] - lse;
}

// ---------------------------------------------------------------------------
static inline size_t align256(size_t x) { return (x + 255) & ~(size_t)255; }

extern "C" void kernel_launch(void* const* d_in, const int* in_sizes, int n_in,
                              void* d_out, int out_size, void* d_ws, size_t ws_size,
                              hipStream_t stream) {
    const float*     x   = (const float*)d_in[0];
    const long long* ei  = (const long long*)d_in[1];   // [2, E] int64
    const float*     W1l = (const float*)d_in[2];
    const float*     b1  = (const float*)d_in[3];
    const float*     W1r = (const float*)d_in[4];
    const float*     W2l = (const float*)d_in[5];
    const float*     b2  = (const float*)d_in[6];
    const float*     W2r = (const float*)d_in[7];
    float*           out = (float*)d_out;

    const int       N = in_sizes[0] / 128;
    const long long E = (long long)in_sizes[1] / 2;

    char* w = (char*)d_ws;
    float*  agg    = (float*)w;  w += align256((size_t)N * 128 * sizeof(float));
    float*  cnt    = (float*)w;  w += align256((size_t)N * sizeof(float));
    float*  h      = (float*)w;  w += align256((size_t)N * 128 * sizeof(float));
    __bf16* W1l_bf = (__bf16*)w; w += align256((size_t)128 * 128 * 2);
    __bf16* W1r_bf = (__bf16*)w; w += align256((size_t)128 * 128 * 2);
    __bf16* W2l_bf = (__bf16*)w; w += align256((size_t)48 * 128 * 2);
    __bf16* W2r_bf = (__bf16*)w; w += align256((size_t)48 * 128 * 2);

    const long long nAgg = (long long)N * 128;
    dim3 zgridAgg((unsigned)((nAgg / 4 + 255) / 256));
    dim3 zgridCnt((unsigned)(((long long)N / 4 + 255) / 256));
    dim3 sgrid((unsigned)((E + 7) / 8));
    dim3 ggrid((unsigned)((N + 127) / 128));
    dim3 fgrid((unsigned)((N + 255) / 256));

    prep_weights<<<64, 256, 0, stream>>>(W1l, W1r, W2l, W2r,
                                         W1l_bf, W1r_bf, W2l_bf, W2r_bf);

    // ---- layer 1 ----
    zero_kernel<<<zgridAgg, 256, 0, stream>>>(agg, nAgg);
    zero_kernel<<<zgridCnt, 256, 0, stream>>>(cnt, N);
    scatter_kernel<<<sgrid, 256, 0, stream>>>(x, ei, E, agg, cnt, 1);
    gemm1_kernel<<<ggrid, 256, 0, stream>>>(agg, cnt, x, W1l_bf, W1r_bf, b1, h, N);

    // ---- layer 2 (reuse agg; counts unchanged) ----
    zero_kernel<<<zgridAgg, 256, 0, stream>>>(agg, nAgg);
    scatter_kernel<<<sgrid, 256, 0, stream>>>(h, ei, E, agg, cnt, 0);
    gemm2_kernel<<<ggrid, 96, 0, stream>>>(agg, cnt, h, W2l_bf, W2r_bf, b2, out, N);

    // ---- normalize + log_softmax ----
    finalize_kernel<<<fgrid, 256, 0, stream>>>(out, N);
}
